// GeometryAwareTensorAttention_68015102099575
// MI455X (gfx1250) — matrix operand order repaired and checked by reading, not compiled, not run on hardware
//
#include <hip/hip_runtime.h>
#include <hip/hip_bf16.h>
#include <cstdint>

// ---------------------------------------------------------------------------
// GeometryAwareTensorAttention for MI455X (gfx1250), wave32 + WMMA f16.
// Heavy math (t_ij@We fused with *pav then @Wc) on v_wmma_f32_16x16x32_f16.
// All weight B-operands are pre-packed into WMMA fragment order so each lane
// loads its 16-half fragment with one contiguous 32-byte vector load.
// ---------------------------------------------------------------------------

typedef __attribute__((ext_vector_type(16))) _Float16 v16h;
typedef __attribute__((ext_vector_type(8)))  _Float16 v8h;
typedef __attribute__((ext_vector_type(8)))  float    v8f;
typedef __attribute__((ext_vector_type(4)))  float    v4f;

#define NN    128      // N
#define DIMC  64       // DIM
#define DIC   512      // DI
#define SC    5        // S
#define HEADS_C 8

__device__ __forceinline__ float fast_rcp(float x) {
    return __builtin_amdgcn_rcpf(x);
}

// Build a 16-half A fragment from two contiguous 8-float runs (f32 source).
//   a[idx] = src[kA(idx)],  kA: idx -> 16*half+idx
__device__ __forceinline__ v16h load_a_f32(const float* __restrict__ row,
                                           int k0, int half) {
    const float* p0 = row + k0 + 16 * half;       // K = k0+16*half .. +7
    const float* p1 = row + k0 + 16 * half + 8;   // K = k0+16*half+8 .. +7
    v4f f0 = *(const v4f*)p0, f1 = *(const v4f*)(p0 + 4);
    v4f f2 = *(const v4f*)p1, f3 = *(const v4f*)(p1 + 4);
    v16h a;
#pragma unroll
    for (int e = 0; e < 4; ++e) {
        a[e]      = (_Float16)f0[e];
        a[4 + e]  = (_Float16)f1[e];
        a[8 + e]  = (_Float16)f2[e];
        a[12 + e] = (_Float16)f3[e];
    }
    return a;
}

// Same but from an f16 LDS row (two ds_load_b128).
__device__ __forceinline__ v16h load_a_f16(const _Float16* __restrict__ row,
                                           int k0, int half) {
    v8h lo = *(const v8h*)(row + k0 + 16 * half);
    v8h hi = *(const v8h*)(row + k0 + 16 * half + 8);
    return __builtin_shufflevector(lo, hi, 0, 1, 2, 3, 4, 5, 6, 7,
                                   8, 9, 10, 11, 12, 13, 14, 15);
}

// ---------------------------------------------------------------------------
// Pack B (KxN row-major f32) into WMMA fragment order (f16):
//  Bp[((kb*NT + nt)*32 + lane)*16 + idx] = B[(kb*32 + 16*(lane>>4) + idx)*N + nt*16 + (lane&15)]
__global__ void packB_kernel(const float* __restrict__ W,
                             _Float16* __restrict__ Wp, int K, int N) {
    int p = blockIdx.x * blockDim.x + threadIdx.x;
    if (p >= K * N) return;
    int idx  = p & 15;
    int lane = (p >> 4) & 31;
    int tile = p >> 9;
    int NT = N >> 4;
    int nt = tile % NT;
    int kb = tile / NT;
    int kk = kb * 32 + 16 * (lane >> 4) + idx;
    int n  = nt * 16 + (lane & 15);
    Wp[p] = (_Float16)W[(size_t)kk * N + n];
}

// Fetch one packed B fragment (32 bytes, coalesced across the wave)
__device__ __forceinline__ v16h load_b_packed(const _Float16* __restrict__ Wp,
                                              int tile, int lane) {
    return *(const v16h*)(Wp + ((size_t)tile * 32 + lane) * 16);
}

// ---------------------------------------------------------------------------
// LayerNorm producing hi and hj (128 rows x 64)
__global__ __launch_bounds__(64)
void ln_kernel(const float* __restrict__ h, const float* __restrict__ wi,
               const float* __restrict__ wj, float* __restrict__ hi,
               float* __restrict__ hj) {
    int i = blockIdx.x, t = threadIdx.x;
    __shared__ float red[64];
    float x = h[i * DIMC + t];
    red[t] = x;
    __syncthreads();
    for (int off = 32; off > 0; off >>= 1) {
        if (t < off) red[t] += red[t + off];
        __syncthreads();
    }
    float mean = red[0] * (1.0f / DIMC);
    __syncthreads();
    float d = x - mean;
    red[t] = d * d;
    __syncthreads();
    for (int off = 32; off > 0; off >>= 1) {
        if (t < off) red[t] += red[t + off];
        __syncthreads();
    }
    float inv = rsqrtf(red[0] * (1.0f / DIMC) + 1e-5f);
    float nrm = d * inv;
    hi[i * DIMC + t] = nrm * wi[t];
    hj[i * DIMC + t] = nrm * wj[t];
}

// ---------------------------------------------------------------------------
// tsum[i,c] = sum_j t_ij[i,j,c]
__global__ __launch_bounds__(64)
void tsum_kernel(const float* __restrict__ t_ij, float* __restrict__ tsum) {
    int i = blockIdx.x, c = threadIdx.x;
    float acc = 0.f;
    for (int j = 0; j < NN; ++j)
        acc += t_ij[((size_t)(i * NN + j)) * DIMC + c];
    tsum[i * DIMC + c] = acc;
}

// ---------------------------------------------------------------------------
// Generic one-wave-per-16x16-tile GEMM: C = act(A(f32) @ Bp(packed f16) + bias)
__global__ __launch_bounds__(32)
void gemm_a32_b16(const float* __restrict__ A, int lda,
                  const _Float16* __restrict__ Bp, int N,
                  const float* __restrict__ bias,
                  float* __restrict__ C, int K, int act) {
    const int n0 = blockIdx.x * 16, m0 = blockIdx.y * 16;
    const int lane = threadIdx.x & 31;
    const int half = lane >> 4, l15 = lane & 15;
    const int NT = N >> 4, nt = blockIdx.x;
    const float* Arow = A + (size_t)(m0 + l15) * lda;
    v8f acc = {0.f, 0.f, 0.f, 0.f, 0.f, 0.f, 0.f, 0.f};
#pragma unroll 2
    for (int k0 = 0; k0 < K; k0 += 32) {
        v16h a = load_a_f32(Arow, k0, half);
        v16h b = load_b_packed(Bp, (k0 >> 5) * NT + nt, lane);
        acc = __builtin_amdgcn_wmma_f32_16x16x32_f16(false, a, false, b,
                                                     (short)0, acc, false, false);
    }
#pragma unroll
    for (int r = 0; r < 8; ++r) {
        int m = m0 + r + 8 * half, n = n0 + l15;
        float v = acc[r];
        if (bias) v += bias[n];
        if (act == 1) v = v * fast_rcp(1.f + __expf(-v));  // SiLU
        C[(size_t)m * N + n] = v;
    }
}

// ---------------------------------------------------------------------------
// sim_num[h,i,j] = sum_d q[i,h*64+d] * k[j,h*64+d]
__global__ __launch_bounds__(32)
void qk_kernel(const float* __restrict__ q, const float* __restrict__ k,
               float* __restrict__ sim_num) {
    const int jt = blockIdx.x, it = blockIdx.y, h = blockIdx.z;
    const int lane = threadIdx.x & 31;
    const int half = lane >> 4, l15 = lane & 15;
    const float* Qrow = q + (size_t)(it * 16 + l15) * DIC + h * 64;
    const float* Krow = k + (size_t)(jt * 16 + l15) * DIC + h * 64;
    v8f acc = {0.f, 0.f, 0.f, 0.f, 0.f, 0.f, 0.f, 0.f};
#pragma unroll
    for (int k0 = 0; k0 < 64; k0 += 32) {
        v16h a = load_a_f32(Qrow, k0, half);
        // B[K=d][N=j] = k[j, h*64+d]; idx runs contiguous in d
        const float* Kp = Krow + k0 + 16 * half;
        v4f g0 = *(const v4f*)Kp, g1 = *(const v4f*)(Kp + 4);
        v4f g2 = *(const v4f*)(Kp + 8), g3 = *(const v4f*)(Kp + 12);
        v16h b;
#pragma unroll
        for (int e = 0; e < 4; ++e) {
            b[e]      = (_Float16)g0[e];
            b[4 + e]  = (_Float16)g1[e];
            b[8 + e]  = (_Float16)g2[e];
            b[12 + e] = (_Float16)g3[e];
        }
        acc = __builtin_amdgcn_wmma_f32_16x16x32_f16(false, a, false, b,
                                                     (short)0, acc, false, false);
    }
#pragma unroll
    for (int r = 0; r < 8; ++r) {
        int ii = it * 16 + r + 8 * half, jj = jt * 16 + l15;
        sim_num[((size_t)(h * NN + ii)) * NN + jj] = acc[r];
    }
}

// ---------------------------------------------------------------------------
// iden[h,i,s] = 1 / sum_d q[i,h*64+d]*esum[i, s*512+h*64+d]
__global__ __launch_bounds__(64)
void den_kernel(const float* __restrict__ q, const float* __restrict__ esum,
                float* __restrict__ iden) {
    int i = blockIdx.x, t = threadIdx.x;
    if (t < HEADS_C * SC) {
        int hh = t / SC, s = t % SC;
        float acc = 0.f;
        for (int d = 0; d < 64; ++d)
            acc += q[i * DIC + hh * 64 + d] *
                   esum[(size_t)i * (SC * DIC) + s * DIC + hh * 64 + d];
        iden[(hh * NN + i) * SC + s] = fast_rcp(acc);
    }
}

// ---------------------------------------------------------------------------
// vc[j,s,m,h] = sum_d v[j, s*512+h*64+d] * Wc[h*64+d, m]   (h innermost!)
__global__ __launch_bounds__(64)
void vc_kernel(const float* __restrict__ vmat, const float* __restrict__ Wc,
               float* __restrict__ vc) {
    int j = blockIdx.x, s = blockIdx.y, m = threadIdx.x;
    float acc[HEADS_C];
#pragma unroll
    for (int hh = 0; hh < HEADS_C; ++hh) {
        float a = 0.f;
        for (int d = 0; d < 64; ++d)
            a += vmat[(size_t)j * (SC * DIC) + s * DIC + hh * 64 + d] *
                 Wc[(hh * 64 + d) * DIMC + m];
        acc[hh] = a;
    }
    float* dst = vc + (((size_t)j * SC + s) * DIMC + m) * HEADS_C;
#pragma unroll
    for (int hh = 0; hh < HEADS_C; ++hh) dst[hh] = acc[hh];
}

// ---------------------------------------------------------------------------
// MAIN fused kernel
__global__ __launch_bounds__(256)
void main_kernel(const float* __restrict__ t_ij,
                 const _Float16* __restrict__ Wep,   // We packed, N=2560 (NT=160)
                 const _Float16* __restrict__ Wcp,   // Wc packed, N=64   (NT=4)
                 const float* __restrict__ pavm,
                 const float* __restrict__ sim_num,
                 const float* __restrict__ iden,
                 const float* __restrict__ vc,
                 float* __restrict__ out_ws) {
    const int i = blockIdx.x;
    const int j0 = blockIdx.y * 32;
    const int tid = threadIdx.x;
    const int wave = tid >> 5;
    const int lane = tid & 31;
    const int half = lane >> 4, l15 = lane & 15;

    __shared__ __align__(16) _Float16 Tt[32 * 64];   //  4 KB
    __shared__ __align__(16) _Float16 EP[32 * 512];  // 32 KB
    __shared__ float sn_s[8 * 32];
    __shared__ float iden_s[8 * 5];

    {   // stage T tile as f16 (vectorized: 8 halves per thread)
        int e8 = tid * 8;
        int jr = e8 >> 6, c = e8 & 63;
        const float* src = &t_ij[((size_t)(i * NN + j0 + jr)) * DIMC + c];
        v4f f0 = *(const v4f*)src, f1 = *(const v4f*)(src + 4);
        v8h o;
#pragma unroll
        for (int e = 0; e < 4; ++e) {
            o[e] = (_Float16)f0[e];
            o[4 + e] = (_Float16)f1[e];
        }
        *(v8h*)&Tt[e8] = o;
    }
    if (tid < 8 * 32) {
        int hh = tid >> 5, jr = tid & 31;
        sn_s[tid] = sim_num[((size_t)(hh * NN + i)) * NN + j0 + jr];
    }
    if (tid >= 256 - 40) {
        int t = tid - (256 - 40);
        iden_s[t] = iden[((t / SC) * NN + i) * SC + (t % SC)];
    }
    __syncthreads();

    for (int s = 0; s < SC; ++s) {
        // ---- GEMM1: E = T(32x64) @ We_s(64x512); epilogue *pav -> EP (f16 LDS)
        for (int t = wave; t < 64; t += 8) {
            int jj = t >> 5;
            int nt = t & 31;
            const _Float16* Arow = &Tt[(jj * 16 + l15) * 64];
            v8f acc = {0.f, 0.f, 0.f, 0.f, 0.f, 0.f, 0.f, 0.f};
#pragma unroll
            for (int k0 = 0; k0 < 64; k0 += 32) {
                v16h a = load_a_f16(Arow, k0, half);
                v16h b = load_b_packed(Wep, (k0 >> 5) * 160 + s * 32 + nt, lane);
                acc = __builtin_amdgcn_wmma_f32_16x16x32_f16(
                    false, a, false, b, (short)0, acc, false, false);
            }
#pragma unroll
            for (int r = 0; r < 8; ++r) {
                int jr = jj * 16 + r + 8 * half;
                int col = nt * 16 + l15;
                float pv = pavm[(size_t)(j0 + jr) * (SC * DIC) + s * DIC + col];
                EP[jr * 512 + col] = (_Float16)(acc[r] * pv);
            }
        }
        __syncthreads();

        // ---- GEMM2: out2 = EP(32x512) @ Wc(512x64) + attn*vc, store
        {
            int jj = wave >> 2;
            int mt = wave & 3;
            const _Float16* Arow = &EP[(jj * 16 + l15) * 512];
            v8f acc = {0.f, 0.f, 0.f, 0.f, 0.f, 0.f, 0.f, 0.f};
#pragma unroll 4
            for (int k0 = 0; k0 < 512; k0 += 32) {
                v16h a = load_a_f16(Arow, k0, half);
                v16h b = load_b_packed(Wcp, (k0 >> 5) * 4 + mt, lane);
                acc = __builtin_amdgcn_wmma_f32_16x16x32_f16(
                    false, a, false, b, (short)0, acc, false, false);
            }
#pragma unroll
            for (int r = 0; r < 8; ++r) {
                int jr = jj * 16 + r + 8 * half;
                int m = mt * 16 + l15;
                float o = acc[r];
                const float* vcb =
                    vc + (((size_t)(j0 + jr) * SC + s) * DIMC + m) * HEADS_C;
                v4f va = *(const v4f*)vcb;
                v4f vb = *(const v4f*)(vcb + 4);
#pragma unroll
                for (int hh = 0; hh < 4; ++hh) {
                    o += (sn_s[hh * 32 + jr] * iden_s[hh * SC + s]) * va[hh];
                    o += (sn_s[(hh + 4) * 32 + jr] * iden_s[(hh + 4) * SC + s]) * vb[hh];
                }
                out_ws[(((size_t)(i * NN) + j0 + jr) * SC + s) * DIMC + m] = o;
            }
        }
        __syncthreads();
    }
}

// ---------------------------------------------------------------------------
// Reductions over j
__global__ __launch_bounds__(64)
void red_h_kernel(const float* __restrict__ h, const float* __restrict__ out_ws,
                  float* __restrict__ h_res) {
    int i = blockIdx.x, m = threadIdx.x;
    float acc = h[i * DIMC + m];
    for (int j = 0; j < NN; ++j)
        acc += out_ws[(((size_t)(i * NN + j)) * SC + 0) * DIMC + m];
    h_res[i * DIMC + m] = acc;
}

__global__ __launch_bounds__(192)
void red_x1_kernel(const float* __restrict__ x1, const float* __restrict__ r_ij1,
                   const float* __restrict__ out_ws, float* __restrict__ x1_res) {
    int i = blockIdx.x, t = threadIdx.x;
    int d = t / 3, m = t % 3;
    float acc = 0.f;
    for (int j = 0; j < NN; ++j) {
        float xs = out_ws[(((size_t)(i * NN + j)) * SC + 3) * DIMC + d];
        float rs = out_ws[(((size_t)(i * NN + j)) * SC + 1) * DIMC + d];
        acc += x1[(j * DIMC + d) * 3 + m] * xs +
               r_ij1[((size_t)(i * NN + j)) * 3 + m] * rs;
    }
    x1_res[(i * DIMC + d) * 3 + m] = acc;
}

__global__ __launch_bounds__(320)
void red_x2_kernel(const float* __restrict__ x2, const float* __restrict__ r_ij2,
                   const float* __restrict__ out_ws, float* __restrict__ x2_res) {
    int i = blockIdx.x, t = threadIdx.x;
    int d = t / 5, m = t % 5;
    float acc = 0.f;
    for (int j = 0; j < NN; ++j) {
        float xs = out_ws[(((size_t)(i * NN + j)) * SC + 4) * DIMC + d];
        float rs = out_ws[(((size_t)(i * NN + j)) * SC + 2) * DIMC + d];
        acc += x2[(j * DIMC + d) * 5 + m] * xs +
               r_ij2[((size_t)(i * NN + j)) * 5 + m] * rs;
    }
    x2_res[(i * DIMC + d) * 5 + m] = acc;
}

// ---------------------------------------------------------------------------
extern "C" void kernel_launch(void* const* d_in, const int* in_sizes, int n_in,
                              void* d_out, int out_size, void* d_ws, size_t ws_size,
                              hipStream_t stream) {
    const float* h     = (const float*)d_in[0];
    const float* x1    = (const float*)d_in[1];
    const float* x2    = (const float*)d_in[2];
    const float* t_ij  = (const float*)d_in[3];
    const float* r_ij1 = (const float*)d_in[4];
    const float* r_ij2 = (const float*)d_in[5];
    const float* wlni  = (const float*)d_in[6];
    const float* wlnj  = (const float*)d_in[7];
    const float* Wq    = (const float*)d_in[8];
    const float* Wk    = (const float*)d_in[9];
    const float* Wv1   = (const float*)d_in[10];
    const float* bv1   = (const float*)d_in[11];
    const float* Wv2   = (const float*)d_in[12];
    const float* bv2   = (const float*)d_in[13];
    const float* Wp1   = (const float*)d_in[14];
    const float* bp1   = (const float*)d_in[15];
    const float* Wp2   = (const float*)d_in[16];
    const float* bp2   = (const float*)d_in[17];
    const float* We    = (const float*)d_in[18];
    const float* Wc    = (const float*)d_in[19];

    char* p = (char*)d_ws;
    auto carve = [&](size_t bytes) -> void* {
        void* r = (void*)p;
        p += (bytes + 255) & ~(size_t)255;
        return r;
    };
    float* hi      = (float*)carve(128 * 64 * 4);
    float* hj      = (float*)carve(128 * 64 * 4);
    float* tsum    = (float*)carve(128 * 64 * 4);
    float* q       = (float*)carve(128 * 512 * 4);
    float* k       = (float*)carve(128 * 512 * 4);
    float* t1      = (float*)carve(128 * 1024 * 4);
    float* t2      = (float*)carve(128 * 1024 * 4);
    float* vmat    = (float*)carve(128 * 2560 * 4);
    float* pavm    = (float*)carve(128 * 2560 * 4);
    float* esum    = (float*)carve(128 * 2560 * 4);
    float* sim_num = (float*)carve(8 * 128 * 128 * 4);
    float* iden    = (float*)carve(8 * 128 * 5 * 4);
    float* vc      = (float*)carve((size_t)128 * 5 * 64 * 8 * 4);
    float* out_ws  = (float*)carve((size_t)128 * 128 * 5 * 64 * 4);
    _Float16* Wq_p  = (_Float16*)carve(64 * 512 * 2);
    _Float16* Wk_p  = (_Float16*)carve(64 * 512 * 2);
    _Float16* Wv1_p = (_Float16*)carve(64 * 1024 * 2);
    _Float16* Wp1_p = (_Float16*)carve(64 * 1024 * 2);
    _Float16* Wv2_p = (_Float16*)carve((size_t)1024 * 2560 * 2);
    _Float16* Wp2_p = (_Float16*)carve((size_t)1024 * 2560 * 2);
    _Float16* We_p  = (_Float16*)carve(64 * 2560 * 2);
    _Float16* Wc_p  = (_Float16*)carve(512 * 64 * 2);

    auto pack = [&](const float* s, _Float16* d, int K, int N) {
        int n = K * N;
        packB_kernel<<<(n + 255) / 256, 256, 0, stream>>>(s, d, K, N);
    };
    pack(Wq, Wq_p, 64, 512);
    pack(Wk, Wk_p, 64, 512);
    pack(Wv1, Wv1_p, 64, 1024);
    pack(Wp1, Wp1_p, 64, 1024);
    pack(Wv2, Wv2_p, 1024, 2560);
    pack(Wp2, Wp2_p, 1024, 2560);
    pack(We, We_p, 64, 2560);
    pack(Wc, Wc_p, 512, 64);

    ln_kernel<<<128, 64, 0, stream>>>(h, wlni, wlnj, hi, hj);
    tsum_kernel<<<128, 64, 0, stream>>>(t_ij, tsum);

    gemm_a32_b16<<<dim3(32, 8), 32, 0, stream>>>(hi, 64, Wq_p, 512, nullptr, q, 64, 0);
    gemm_a32_b16<<<dim3(32, 8), 32, 0, stream>>>(hj, 64, Wk_p, 512, nullptr, k, 64, 0);
    gemm_a32_b16<<<dim3(64, 8), 32, 0, stream>>>(hj, 64, Wv1_p, 1024, bv1, t1, 64, 1);
    gemm_a32_b16<<<dim3(160, 8), 32, 0, stream>>>(t1, 1024, Wv2_p, 2560, bv2, vmat, 1024, 0);
    gemm_a32_b16<<<dim3(64, 8), 32, 0, stream>>>(hj, 64, Wp1_p, 1024, bp1, t2, 64, 1);
    gemm_a32_b16<<<dim3(160, 8), 32, 0, stream>>>(t2, 1024, Wp2_p, 2560, bp2, pavm, 1024, 0);
    gemm_a32_b16<<<dim3(160, 8), 32, 0, stream>>>(tsum, 64, We_p, 2560, nullptr, esum, 64, 0);

    qk_kernel<<<dim3(8, 8, 8), 32, 0, stream>>>(q, k, sim_num);
    den_kernel<<<128, 64, 0, stream>>>(q, esum, iden);
    vc_kernel<<<dim3(128, 5), 64, 0, stream>>>(vmat, Wc, vc);

    main_kernel<<<dim3(128, 4), 256, 0, stream>>>(t_ij, We_p, Wc_p, pavm,
                                                  sim_num, iden, vc, out_ws);

    float* h_res  = (float*)d_out;
    float* x1_res = (float*)d_out + 8192;
    float* x2_res = (float*)d_out + 32768;
    red_h_kernel<<<128, 64, 0, stream>>>(h, out_ws, h_res);
    red_x1_kernel<<<128, 192, 0, stream>>>(x1, r_ij1, out_ws, x1_res);
    red_x2_kernel<<<128, 320, 0, stream>>>(x2, r_ij2, out_ws, x2_res);
}
